// MoEDetect_66073776881831
// MI455X (gfx1250) — matrix operand order repaired and checked by reading, not compiled, not run on hardware
//
#include <hip/hip_runtime.h>

// ---------------------------------------------------------------------------
// MoE detect head for MI455X (gfx1250): bf16 WMMA implicit-GEMM convs + MoE.
// M-tile = 32 pixels -> 2 WMMAs per B-fragment load (B register reuse).
// ---------------------------------------------------------------------------

typedef __attribute__((ext_vector_type(16))) __bf16        v16bf;
typedef __attribute__((ext_vector_type(8)))  float         v8f;
typedef __attribute__((ext_vector_type(4)))  float         v4f;
typedef __attribute__((ext_vector_type(4)))  unsigned int  v4u;
typedef __attribute__((ext_vector_type(2)))  unsigned int  v2u;

union AFrag {
  v16bf v;
  unsigned short u[16];
  v4u q[2];
};

__device__ __forceinline__ unsigned short f32_to_bf16(float f) {
  unsigned int u = __float_as_uint(f);
  u += 0x7FFFu + ((u >> 16) & 1u);          // round-to-nearest-even
  return (unsigned short)(u >> 16);
}
__device__ __forceinline__ float bf16_to_f32(unsigned short h) {
  return __uint_as_float(((unsigned int)h) << 16);
}

// LDS A tile: [M=32][K=32], pitch 40 ushorts (80B): 16B-aligned rows, spread
// across banks; each lane's A-fragment half is one 16B ds_load_b128.
#define APITCH 40

// ---------------------------------------------------------------------------
// Weight repack: conv [L][Cout][Cin][3][3] fp32 -> bf16 [L][tap*Cin+cin][Cout]
// ---------------------------------------------------------------------------
__global__ void repack_conv_w(const float* __restrict__ src,
                              unsigned short* __restrict__ dst,
                              int L, int Cout, int Cin) {
  int per = Cout * Cin * 9;
  int total = L * per;
  int i = blockIdx.x * blockDim.x + threadIdx.x;
  if (i >= total) return;
  int l    = i / per;
  int r    = i % per;
  int cout = r / (Cin * 9);
  int r2   = r % (Cin * 9);
  int cin  = r2 / 9;
  int tap  = r2 % 9;
  dst[l * per + (tap * Cin + cin) * Cout + cout] = f32_to_bf16(src[i]);
}

// MoE: [L][E][Cout][Cin] fp32 -> bf16 [L][E][Cin][Cout]
__global__ void repack_moe_w(const float* __restrict__ src,
                             unsigned short* __restrict__ dst,
                             int L, int E, int Cout, int Cin) {
  int per = E * Cout * Cin;
  int total = L * per;
  int i = blockIdx.x * blockDim.x + threadIdx.x;
  if (i >= total) return;
  int l    = i / per;
  int r    = i % per;
  int e    = r / (Cout * Cin);
  int r2   = r % (Cout * Cin);
  int cout = r2 / Cin;
  int cin  = r2 % Cin;
  dst[l * per + (e * Cin + cin) * Cout + cout] = f32_to_bf16(src[i]);
}

// ---------------------------------------------------------------------------
// Conv3x3 (SAME) + bias + SiLU, implicit GEMM with bf16 WMMA.
// Block = 128 threads (4 waves): 32-pixel M tile x 64 output channels.
// gridDim.x = B*H*ceil(W/32), gridDim.y = Cout/64.
// wp: bf16 [tap*Cin+cin][Cout].  Output: bf16 NCHW.
// ---------------------------------------------------------------------------
template <bool IN_F32>
__global__ __launch_bounds__(128)
void conv3x3_silu(const void* __restrict__ in_,
                  const unsigned short* __restrict__ wp,
                  const float* __restrict__ bias,
                  unsigned short* __restrict__ outp,
                  int Cin, int Cout, int H, int W) {
  __shared__ __align__(16) unsigned short sA[2][32 * APITCH];

  const int tiles_per_row = (W + 31) >> 5;
  const int mt = blockIdx.x;
  const int x0 = (mt % tiles_per_row) << 5;
  const int t  = mt / tiles_per_row;
  const int y  = t % H;
  const int b  = t / H;

  const int lane  = threadIdx.x & 31;
  const int wave  = threadIdx.x >> 5;
  const int nbase = blockIdx.y * 64 + wave * 16;
  const int HW    = H * W;

  const float*          inF = (const float*)in_;
  const unsigned short* inH = (const unsigned short*)in_;

  const int sk  = threadIdx.x & 31;        // channel-in-chunk (K)
  const int sm  = (threadIdx.x >> 5) << 3; // 8-pixel group base (M)
  const int cpr = Cin >> 5;                // K-chunks per tap
  const int nchunks = 9 * cpr;
  const int inb = b * Cin * HW;

  auto stage = [&](int kc, int bufi) {
    int tap = kc / cpr;
    int cc  = (kc - tap * cpr) << 5;
    int dy  = tap / 3 - 1, dx = tap % 3 - 1;
    int yy  = y + dy;
    bool yok = (unsigned)yy < (unsigned)H;
    int base = inb + (cc + sk) * HW + (yok ? yy : 0) * W;
    int xs   = x0 + sm + dx;
    unsigned short* dstp = &sA[bufi][sm * APITCH + sk];
    if (yok && xs >= 0 && (xs + 7) < W) {
      // interior: unconditional loads (clause-able, single wait)
      #pragma unroll
      for (int j = 0; j < 8; ++j) {
        float v = IN_F32 ? inF[base + xs + j] : bf16_to_f32(inH[base + xs + j]);
        dstp[j * APITCH] = f32_to_bf16(v);
      }
    } else {
      #pragma unroll
      for (int j = 0; j < 8; ++j) {
        int xx = xs + j;
        float v = 0.f;
        if (yok && (unsigned)xx < (unsigned)W)
          v = IN_F32 ? inF[base + xx] : bf16_to_f32(inH[base + xx]);
        dstp[j * APITCH] = f32_to_bf16(v);
      }
    }
  };

  v8f acc0 = {0.f, 0.f, 0.f, 0.f, 0.f, 0.f, 0.f, 0.f};
  v8f acc1 = {0.f, 0.f, 0.f, 0.f, 0.f, 0.f, 0.f, 0.f};
  const int M   = lane & 15;
  const int hi8 = (lane >> 4) << 3;

  stage(0, 0);
  int buf = 0;
  for (int kc = 0; kc < nchunks; ++kc) {
    __syncthreads();
    if (kc + 1 < nchunks) stage(kc + 1, buf ^ 1);

    const unsigned short* brow = wp + (kc * 32 + hi8) * Cout + nbase + M;
    AFrag bf;
    #pragma unroll
    for (int s = 0; s < 8; ++s) {
      bf.u[s]     = brow[s * Cout];
      bf.u[8 + s] = brow[(16 + s) * Cout];
    }
    v16bf bv = bf.v;
    if (kc + 1 < nchunks) __builtin_prefetch(brow + 32 * Cout, 0, 0);

    AFrag a0, a1;
    const unsigned short* r0 = &sA[buf][M * APITCH];
    const unsigned short* r1 = &sA[buf][(16 + M) * APITCH];
    a0.q[0] = *(const v4u*)(r0 + hi8);        // K = hi8 .. hi8+7
    a0.q[1] = *(const v4u*)(r0 + 16 + hi8);   // K = 16+hi8 .. +7
    a1.q[0] = *(const v4u*)(r1 + hi8);
    a1.q[1] = *(const v4u*)(r1 + 16 + hi8);

    acc0 = __builtin_amdgcn_wmma_f32_16x16x32_bf16(false, a0.v, false, bv,
                                                   (short)0, acc0, false, false);
    acc1 = __builtin_amdgcn_wmma_f32_16x16x32_bf16(false, a1.v, false, bv,
                                                   (short)0, acc1, false, false);
    buf ^= 1;
  }

  // epilogue: bias + SiLU -> bf16 NCHW (two 16-pixel sub-tiles per lane)
  const int n  = nbase + (lane & 15);
  const float bs = bias[n];
  const int orow = (b * Cout + n) * HW + y * W;
  #pragma unroll
  for (int s = 0; s < 2; ++s) {
    v8f acc = s ? acc1 : acc0;
    unsigned short sv[8];
    #pragma unroll
    for (int m = 0; m < 8; ++m) {
      float v = acc[m] + bs;
      v = v / (1.f + __expf(-v));
      sv[m] = f32_to_bf16(v);
    }
    const int px = x0 + s * 16 + hi8;
    if (x0 + s * 16 + 15 < W) {
      v2u p0, p1;
      p0.x = (unsigned)sv[0] | ((unsigned)sv[1] << 16);
      p0.y = (unsigned)sv[2] | ((unsigned)sv[3] << 16);
      p1.x = (unsigned)sv[4] | ((unsigned)sv[5] << 16);
      p1.y = (unsigned)sv[6] | ((unsigned)sv[7] << 16);
      *(v2u*)(outp + orow + px)     = p0;     // 8B aligned
      *(v2u*)(outp + orow + px + 4) = p1;
    } else {
      #pragma unroll
      for (int m = 0; m < 8; ++m)
        if (px + m < W) outp[orow + px + m] = sv[m];
    }
  }
}

// ---------------------------------------------------------------------------
// MoE 1x1: per-sample expert GEMM, same staged-WMMA skeleton (single tap).
// Block = 128 threads (4 waves); 32-pixel M tile; gridDim.y = ceil(Coutm/64);
// waves with nbase >= Coutm only help stage. Output fp32 into [B][144][H][W].
// ---------------------------------------------------------------------------
__global__ __launch_bounds__(128)
void moe1x1(const unsigned short* __restrict__ act,
            const unsigned short* __restrict__ wep,
            const float* __restrict__ bep,
            const int* __restrict__ midx,
            float* __restrict__ outp,
            int Cin, int Coutm, int H, int W, int ch_off, int ch_total) {
  __shared__ __align__(16) unsigned short sA[2][32 * APITCH];

  const int tiles_per_row = (W + 31) >> 5;
  const int mt = blockIdx.x;
  const int x0 = (mt % tiles_per_row) << 5;
  const int t  = mt / tiles_per_row;
  const int y  = t % H;
  const int b  = t / H;

  const int lane  = threadIdx.x & 31;
  const int wave  = threadIdx.x >> 5;
  const int nbase = blockIdx.y * 64 + wave * 16;
  const bool active = nbase < Coutm;
  const int HW = H * W;
  const int e  = midx[b];

  const int sk = threadIdx.x & 31;
  const int sm = (threadIdx.x >> 5) << 3;
  const int abase0 = b * Cin * HW + y * W + x0 + sm;
  const bool fullt = (x0 + 31) < W;       // excludes W=20 misaligned rows too
  const int nchunks = Cin >> 5;

  auto stage = [&](int cc, int bufi) {
    int base = abase0 + (cc + sk) * HW;
    unsigned short* dstp = &sA[bufi][sm * APITCH + sk];
    if (fullt) {
      v4u d = *(const v4u*)(act + base);      // 8 bf16, 16B aligned
      #pragma unroll
      for (int j = 0; j < 4; ++j) {
        dstp[(2 * j) * APITCH]     = (unsigned short)(d[j] & 0xffffu);
        dstp[(2 * j + 1) * APITCH] = (unsigned short)(d[j] >> 16);
      }
    } else {
      #pragma unroll
      for (int j = 0; j < 8; ++j) {
        int xx = x0 + sm + j;
        dstp[j * APITCH] = (xx < W) ? act[base + j] : (unsigned short)0;
      }
    }
  };

  v8f acc0 = {0.f, 0.f, 0.f, 0.f, 0.f, 0.f, 0.f, 0.f};
  v8f acc1 = {0.f, 0.f, 0.f, 0.f, 0.f, 0.f, 0.f, 0.f};
  const int M   = lane & 15;
  const int hi8 = (lane >> 4) << 3;
  const int ebase = e * Cin;

  stage(0, 0);
  int buf = 0;
  for (int kc = 0; kc < nchunks; ++kc) {
    __syncthreads();
    if (kc + 1 < nchunks) stage((kc + 1) << 5, buf ^ 1);
    if (active) {
      const unsigned short* brow =
          wep + (ebase + (kc << 5) + hi8) * Coutm + nbase + M;
      AFrag bf;
      #pragma unroll
      for (int s = 0; s < 8; ++s) {
        bf.u[s]     = brow[s * Coutm];
        bf.u[8 + s] = brow[(16 + s) * Coutm];
      }
      v16bf bv = bf.v;
      AFrag a0, a1;
      const unsigned short* r0 = &sA[buf][M * APITCH];
      const unsigned short* r1 = &sA[buf][(16 + M) * APITCH];
      a0.q[0] = *(const v4u*)(r0 + hi8);
      a0.q[1] = *(const v4u*)(r0 + 16 + hi8);
      a1.q[0] = *(const v4u*)(r1 + hi8);
      a1.q[1] = *(const v4u*)(r1 + 16 + hi8);
      acc0 = __builtin_amdgcn_wmma_f32_16x16x32_bf16(false, a0.v, false, bv,
                                                     (short)0, acc0, false, false);
      acc1 = __builtin_amdgcn_wmma_f32_16x16x32_bf16(false, a1.v, false, bv,
                                                     (short)0, acc1, false, false);
    }
    buf ^= 1;
  }

  if (active) {
    const int n = nbase + (lane & 15);
    const float bs = bep[e * Coutm + n];
    const int orow = (b * ch_total + ch_off + n) * HW + y * W;
    #pragma unroll
    for (int s = 0; s < 2; ++s) {
      v8f acc = s ? acc1 : acc0;
      const int px = x0 + s * 16 + hi8;
      if (x0 + s * 16 + 15 < W) {
        v4f f0, f1;
        #pragma unroll
        for (int m = 0; m < 4; ++m) { f0[m] = acc[m] + bs; f1[m] = acc[m + 4] + bs; }
        *(v4f*)(outp + orow + px)     = f0;   // 16B aligned (W=80/40/20 ok)
        *(v4f*)(outp + orow + px + 4) = f1;
      } else {
        #pragma unroll
        for (int m = 0; m < 8; ++m)
          if (px + m < W) outp[orow + px + m] = acc[m] + bs;
      }
    }
  }
}

// ---------------------------------------------------------------------------
extern "C" void kernel_launch(void* const* d_in, const int* in_sizes, int n_in,
                              void* d_out, int out_size, void* d_ws, size_t ws_size,
                              hipStream_t stream) {
  (void)in_sizes; (void)n_in; (void)out_size; (void)ws_size;

  const float* x[3] = {(const float*)d_in[0], (const float*)d_in[1],
                       (const float*)d_in[2]};
  const int*   midx   = (const int*)d_in[3];
  const float* cv2_w1 = (const float*)d_in[4];
  const float* cv2_b1 = (const float*)d_in[5];
  const float* cv2_w2 = (const float*)d_in[6];
  const float* cv2_b2 = (const float*)d_in[7];
  const float* cv2_we = (const float*)d_in[8];
  const float* cv2_be = (const float*)d_in[9];
  const float* cv3_w1 = (const float*)d_in[10];
  const float* cv3_b1 = (const float*)d_in[11];
  const float* cv3_w2 = (const float*)d_in[12];
  const float* cv3_b2 = (const float*)d_in[13];
  const float* cv3_we = (const float*)d_in[14];
  const float* cv3_be = (const float*)d_in[15];
  float* out = (float*)d_out;

  // ---- workspace carve-up (bf16 elements) ----
  char* ws = (char*)d_ws;
  size_t off = 0;
  auto take = [&](size_t elems) -> unsigned short* {
    unsigned short* p = (unsigned short*)(ws + off);
    off += ((elems * 2) + 255) & ~(size_t)255;
    return p;
  };
  unsigned short* w2a  = take((size_t)3 * 147456);   // cv2_w1 repacked
  unsigned short* w2b  = take((size_t)3 * 36864);    // cv2_w2
  unsigned short* w3a  = take((size_t)3 * 589824);   // cv3_w1
  unsigned short* w3b  = take((size_t)3 * 589824);   // cv3_w2
  unsigned short* w2e  = take((size_t)3 * 32768);    // cv2_we
  unsigned short* w3e  = take((size_t)3 * 163840);   // cv3_we
  unsigned short* actA = take((size_t)16 * 256 * 6400);
  unsigned short* actB = take((size_t)16 * 256 * 6400);

  auto cdiv = [](int a, int b) { return (a + b - 1) / b; };

  // ---- weight conversion / repack ----
  int t;
  t = 3 * 147456; repack_conv_w<<<cdiv(t, 256), 256, 0, stream>>>(cv2_w1, w2a, 3, 64, 256);
  t = 3 * 36864;  repack_conv_w<<<cdiv(t, 256), 256, 0, stream>>>(cv2_w2, w2b, 3, 64, 64);
  t = 3 * 589824; repack_conv_w<<<cdiv(t, 256), 256, 0, stream>>>(cv3_w1, w3a, 3, 256, 256);
  t = 3 * 589824; repack_conv_w<<<cdiv(t, 256), 256, 0, stream>>>(cv3_w2, w3b, 3, 256, 256);
  t = 3 * 32768;  repack_moe_w<<<cdiv(t, 256), 256, 0, stream>>>(cv2_we, w2e, 3, 8, 64, 64);
  t = 3 * 163840; repack_moe_w<<<cdiv(t, 256), 256, 0, stream>>>(cv3_we, w3e, 3, 8, 80, 256);

  const int B = 16;
  const int Hs[3] = {80, 40, 20};
  size_t lvl_off = 0;
  for (int l = 0; l < 3; ++l) {
    int H = Hs[l], W = H;
    int mtiles = B * H * cdiv(W, 32);
    // ---- cv2 branch (256->64->64, MoE 64->64) ----
    conv3x3_silu<true><<<dim3(mtiles, 1), 128, 0, stream>>>(
        x[l], w2a + (size_t)l * 147456, cv2_b1 + l * 64, actA, 256, 64, H, W);
    conv3x3_silu<false><<<dim3(mtiles, 1), 128, 0, stream>>>(
        actA, w2b + (size_t)l * 36864, cv2_b2 + l * 64, actB, 64, 64, H, W);
    moe1x1<<<dim3(mtiles, 1), 128, 0, stream>>>(
        actB, w2e + (size_t)l * 32768, cv2_be + (size_t)l * 8 * 64, midx,
        out + lvl_off, 64, 64, H, W, 0, 144);
    // ---- cv3 branch (256->256->256, MoE 256->80) ----
    conv3x3_silu<true><<<dim3(mtiles, 4), 128, 0, stream>>>(
        x[l], w3a + (size_t)l * 589824, cv3_b1 + l * 256, actA, 256, 256, H, W);
    conv3x3_silu<false><<<dim3(mtiles, 4), 128, 0, stream>>>(
        actA, w3b + (size_t)l * 589824, cv3_b2 + l * 256, actB, 256, 256, H, W);
    moe1x1<<<dim3(mtiles, 2), 128, 0, stream>>>(
        actB, w3e + (size_t)l * 163840, cv3_be + (size_t)l * 8 * 80, midx,
        out + lvl_off, 256, 80, H, W, 64, 144);
    lvl_off += (size_t)B * 144 * H * W;
  }
}
